// SigmaMoETransformerLayer_67216238182689
// MI455X (gfx1250) — matrix operand order repaired and checked by reading, not compiled, not run on hardware
//
#include <hip/hip_runtime.h>
#include <hip/hip_bf16.h>
#include <stdint.h>

// ---------------------------------------------------------------------------
// Problem constants (SigmaMoE transformer layer)
// ---------------------------------------------------------------------------
#define BB 2
#define SS 2048
#define DD 1024
#define HH 16
#define DHH 64
#define EE 16
#define FF 256
#define TT (BB * SS)   // 4096 tokens

typedef __attribute__((ext_vector_type(16))) __bf16 v16bf;
typedef __attribute__((ext_vector_type(8)))  float  v8f;

union BF2 { uint32_t u32; __bf16 b[2]; };

// ---------------------------------------------------------------------------
// WMMA helpers (CDNA5 wave32, v_wmma_f32_16x16x32_bf16)
// ---------------------------------------------------------------------------
__device__ __forceinline__ v8f wmma_bf16(v16bf a, v16bf b, v8f c) {
  return __builtin_amdgcn_wmma_f32_16x16x32_bf16(
      false, a, false, b, (short)0, c, false, false);
}

// A fragment: 16x32 (MxK), row-major A, lda in elements
__device__ __forceinline__ v16bf load_a_frag(const __bf16* __restrict__ A, int lda,
                                             int m0, int k0, int lane) {
  int half = lane >> 4;
  int row  = m0 + (lane & 15);
  const __bf16* p = A + (size_t)row * lda + k0;
  v16bf a;
#pragma unroll
  for (int j = 0; j < 8; ++j) {
    int koff = ((j < 4) ? 2 * j : 2 * j + 8) + 8 * half;
    BF2 t; t.u32 = *reinterpret_cast<const uint32_t*>(p + koff);
    a[2 * j] = t.b[0]; a[2 * j + 1] = t.b[1];
  }
  return a;
}

// B fragment: 32x16 (KxN); B stored transposed (BT row-major [N][K])
__device__ __forceinline__ v16bf load_bt_frag(const __bf16* __restrict__ BT, int ldb,
                                              int n0, int k0, int lane) {
  int half = lane >> 4;
  int col  = n0 + (lane & 15);
  const __bf16* p = BT + (size_t)col * ldb + k0;
  v16bf b;
#pragma unroll
  for (int j = 0; j < 8; ++j) {
    int koff = ((j < 4) ? 2 * j : 2 * j + 8) + 8 * half;
    BF2 t; t.u32 = *reinterpret_cast<const uint32_t*>(p + koff);
    b[2 * j] = t.b[0]; b[2 * j + 1] = t.b[1];
  }
  return b;
}

__device__ __forceinline__ float half_reduce_max(float v) {
#pragma unroll
  for (int m = 1; m < 16; m <<= 1) v = fmaxf(v, __shfl_xor(v, m, 32));
  return v;
}
__device__ __forceinline__ float half_reduce_sum(float v) {
#pragma unroll
  for (int m = 1; m < 16; m <<= 1) v += __shfl_xor(v, m, 32);
  return v;
}

// 32x64 register-tile inner product: acc[2][4] over K
__device__ __forceinline__ void gemm_tile_32x64(const __bf16* __restrict__ A, int lda,
                                                const __bf16* __restrict__ BT, int ldb,
                                                int m0, int n0, int kbeg, int kend,
                                                int lane, v8f acc[2][4]) {
  for (int k0 = kbeg; k0 < kend; k0 += 32) {
    v16bf a0 = load_a_frag(A, lda, m0,      k0, lane);
    v16bf a1 = load_a_frag(A, lda, m0 + 16, k0, lane);
    v16bf b0 = load_bt_frag(BT, ldb, n0,      k0, lane);
    v16bf b1 = load_bt_frag(BT, ldb, n0 + 16, k0, lane);
    v16bf b2 = load_bt_frag(BT, ldb, n0 + 32, k0, lane);
    v16bf b3 = load_bt_frag(BT, ldb, n0 + 48, k0, lane);
    acc[0][0] = wmma_bf16(a0, b0, acc[0][0]);
    acc[0][1] = wmma_bf16(a0, b1, acc[0][1]);
    acc[0][2] = wmma_bf16(a0, b2, acc[0][2]);
    acc[0][3] = wmma_bf16(a0, b3, acc[0][3]);
    acc[1][0] = wmma_bf16(a1, b0, acc[1][0]);
    acc[1][1] = wmma_bf16(a1, b1, acc[1][1]);
    acc[1][2] = wmma_bf16(a1, b2, acc[1][2]);
    acc[1][3] = wmma_bf16(a1, b3, acc[1][3]);
  }
}

// ---------------------------------------------------------------------------
// Weight cast+transpose: src fp32 [nmat][R][C] -> dst bf16 [nmat][C][R]
// ---------------------------------------------------------------------------
__global__ void tcast_kernel(const float* __restrict__ src, __bf16* __restrict__ dst,
                             int R, int C, long long total) {
  long long i = (long long)blockIdx.x * 256 + threadIdx.x;
  if (i >= total) return;
  long long rc = (long long)R * C;
  long long mat = i / rc;
  long long rem = i - mat * rc;
  int r = (int)(rem / C);
  int c = (int)(rem - (long long)r * C);
  dst[mat * rc + (long long)c * R + r] = (__bf16)src[i];
}

// ---------------------------------------------------------------------------
// LayerNorm (fp32 in -> bf16 out), one block per token row, D = 1024
// ---------------------------------------------------------------------------
__global__ void ln_bf16_kernel(const float* __restrict__ x, const float* __restrict__ sc,
                               const float* __restrict__ bi, __bf16* __restrict__ out) {
  int row = blockIdx.x, tid = threadIdx.x;
  const float* xr = x + (size_t)row * DD;
  float s = 0.f, sq = 0.f;
  for (int i = tid; i < DD; i += 256) { float v = xr[i]; s += v; sq += v * v; }
  __shared__ float sh_s[256], sh_q[256];
  sh_s[tid] = s; sh_q[tid] = sq; __syncthreads();
  for (int off = 128; off > 0; off >>= 1) {
    if (tid < off) { sh_s[tid] += sh_s[tid + off]; sh_q[tid] += sh_q[tid + off]; }
    __syncthreads();
  }
  float mu = sh_s[0] * (1.f / DD);
  float var = sh_q[0] * (1.f / DD) - mu * mu;
  float rstd = rsqrtf(var + 1e-5f);
  for (int i = tid; i < DD; i += 256) {
    float v = (xr[i] - mu) * rstd * sc[i] + bi[i];
    out[(size_t)row * DD + i] = (__bf16)v;
  }
}

// ---------------------------------------------------------------------------
// GEMM (bf16 out): C[M,N] = scale * A @ BT^T.  Wave tile 32x64.
// grid = (M/32, N/256), block = 128 (4 waves)
// ---------------------------------------------------------------------------
__global__ void gemm_bf16_out_kernel(const __bf16* __restrict__ A, const __bf16* __restrict__ BT,
                                     __bf16* __restrict__ C, int N, int K, float scale) {
  int lane = threadIdx.x & 31, wave = threadIdx.x >> 5;
  int m0 = blockIdx.x * 32;
  int n0 = blockIdx.y * 256 + wave * 64;
  v8f acc[2][4] = {};
  gemm_tile_32x64(A, K, BT, K, m0, n0, 0, K, lane, acc);
  int half = lane >> 4, ln = lane & 15;
#pragma unroll
  for (int mi = 0; mi < 2; ++mi)
#pragma unroll
    for (int ni = 0; ni < 4; ++ni) {
      int col = n0 + ni * 16 + ln;
#pragma unroll
      for (int r = 0; r < 8; ++r) {
        int m = m0 + mi * 16 + 8 * half + r;
        C[(size_t)m * N + col] = (__bf16)(acc[mi][ni][r] * scale);
      }
    }
}

// V projection with per-head transposed store: vT[b][h][dh][s]
__global__ void gemm_vT_kernel(const __bf16* __restrict__ A, const __bf16* __restrict__ BT,
                               __bf16* __restrict__ vT, int N, int K) {
  int lane = threadIdx.x & 31, wave = threadIdx.x >> 5;
  int m0 = blockIdx.x * 32;
  int n0 = blockIdx.y * 256 + wave * 64;
  v8f acc[2][4] = {};
  gemm_tile_32x64(A, K, BT, K, m0, n0, 0, K, lane, acc);
  int half = lane >> 4, ln = lane & 15;
#pragma unroll
  for (int mi = 0; mi < 2; ++mi)
#pragma unroll
    for (int ni = 0; ni < 4; ++ni) {
      int col = n0 + ni * 16 + ln;       // global column = h*64 + dh
      int h  = col >> 6, dh = col & 63;
      int mb = m0 + mi * 16 + 8 * half;  // 8 consecutive tokens
      int b  = mb >> 11;                 // SS = 2048
      int s  = mb & (SS - 1);
      __bf16* dst = vT + (((size_t)b * HH + h) * DHH + dh) * SS + s;
#pragma unroll
      for (int r = 0; r < 8; ++r) dst[r] = (__bf16)acc[mi][ni][r];  // contiguous -> b128
    }
}

// out_f32[M,N] = res_f32 + A @ BT^T   (Wo projection + residual)
__global__ void gemm_res_f32_kernel(const __bf16* __restrict__ A, const __bf16* __restrict__ BT,
                                    const float* __restrict__ res, float* __restrict__ out,
                                    int N, int K) {
  int lane = threadIdx.x & 31, wave = threadIdx.x >> 5;
  int m0 = blockIdx.x * 32;
  int n0 = blockIdx.y * 256 + wave * 64;
  v8f acc[2][4] = {};
  gemm_tile_32x64(A, K, BT, K, m0, n0, 0, K, lane, acc);
  int half = lane >> 4, ln = lane & 15;
#pragma unroll
  for (int mi = 0; mi < 2; ++mi)
#pragma unroll
    for (int ni = 0; ni < 4; ++ni) {
      int col = n0 + ni * 16 + ln;
#pragma unroll
      for (int r = 0; r < 8; ++r) {
        int m = m0 + mi * 16 + 8 * half + r;
        out[(size_t)m * N + col] = res[(size_t)m * N + col] + acc[mi][ni][r];
      }
    }
}

// ---------------------------------------------------------------------------
// Flash attention: one wave = one 16-query tile of one (b,h).
// q pre-scaled by 1/sqrt(DH). V consumed from per-head transposed vT.
// grid = (S/64, B*H), block = 128.
// ---------------------------------------------------------------------------
__global__ void attn_kernel(const __bf16* __restrict__ qb, const __bf16* __restrict__ kb,
                            const __bf16* __restrict__ vT, __bf16* __restrict__ ctxb) {
  int lane = threadIdx.x & 31, wave = threadIdx.x >> 5;
  int bh = blockIdx.y;
  int b = bh >> 4, h = bh & 15;
  int q0 = (blockIdx.x * 4 + wave) * 16;
  size_t off  = (size_t)b * SS * DD + (size_t)h * DHH;        // q/k/ctx layout [T,D]
  const __bf16* vh = vT + ((size_t)b * HH + h) * DHH * SS;    // [DH][S]
  __shared__ __bf16 pshm[4][16 * 32];
  __bf16* ps = pshm[wave];
  int half = lane >> 4, ln = lane & 15;

  v16bf aq0 = load_a_frag(qb + off, DD, q0, 0, lane);
  v16bf aq1 = load_a_frag(qb + off, DD, q0, 32, lane);
  v8f acc0 = {}, acc1 = {}, acc2 = {}, acc3 = {};
  float mrow[8], lrow[8];
#pragma unroll
  for (int r = 0; r < 8; ++r) { mrow[r] = -3.0e38f; lrow[r] = 0.f; }

  for (int key0 = 0; key0 < SS; key0 += 32) {
    v8f s0 = {}, s1 = {};
    s0 = wmma_bf16(aq0, load_bt_frag(kb + off, DD, key0, 0, lane), s0);
    s0 = wmma_bf16(aq1, load_bt_frag(kb + off, DD, key0, 32, lane), s0);
    s1 = wmma_bf16(aq0, load_bt_frag(kb + off, DD, key0 + 16, 0, lane), s1);
    s1 = wmma_bf16(aq1, load_bt_frag(kb + off, DD, key0 + 16, 32, lane), s1);
#pragma unroll
    for (int r = 0; r < 8; ++r) {
      float mx   = half_reduce_max(fmaxf(s0[r], s1[r]));
      float mnew = fmaxf(mrow[r], mx);
      float scl  = __expf(mrow[r] - mnew);
      float p0   = __expf(s0[r] - mnew);
      float p1   = __expf(s1[r] - mnew);
      float rs   = half_reduce_sum(p0 + p1);
      lrow[r] = lrow[r] * scl + rs;
      mrow[r] = mnew;
      acc0[r] *= scl; acc1[r] *= scl; acc2[r] *= scl; acc3[r] *= scl;
      int m = r + 8 * half;
      ps[m * 32 + ln]      = (__bf16)p0;
      ps[m * 32 + 16 + ln] = (__bf16)p1;
    }
    asm volatile("s_wait_dscnt 0" ::: "memory");  // P tile in LDS before re-read
    v16bf ap = load_a_frag(ps, 32, 0, 0, lane);
    // P(16x32) x V(32x64): vh is [DH][S] => BT layout with ldb = S
    acc0 = wmma_bf16(ap, load_bt_frag(vh, SS, 0,  key0, lane), acc0);
    acc1 = wmma_bf16(ap, load_bt_frag(vh, SS, 16, key0, lane), acc1);
    acc2 = wmma_bf16(ap, load_bt_frag(vh, SS, 32, key0, lane), acc2);
    acc3 = wmma_bf16(ap, load_bt_frag(vh, SS, 48, key0, lane), acc3);
  }
#pragma unroll
  for (int r = 0; r < 8; ++r) {
    int m = r + 8 * half;
    float inv = 1.f / lrow[r];
    size_t ro = off + (size_t)(q0 + m) * DD;
    ctxb[ro + 0  + ln] = (__bf16)(acc0[r] * inv);
    ctxb[ro + 16 + ln] = (__bf16)(acc1[r] * inv);
    ctxb[ro + 32 + ln] = (__bf16)(acc2[r] * inv);
    ctxb[ro + 48 + ln] = (__bf16)(acc3[r] * inv);
  }
}

// ---------------------------------------------------------------------------
// Router: sigmoid(h2 @ W_sel) top-2 -> gates + 32-token-tile expert flags
// block = 256 (8 waves), one wave per token. WselT is bf16 [E][D].
// ---------------------------------------------------------------------------
__global__ void zero_i32_kernel(int* __restrict__ p, int n) {
  int i = blockIdx.x * 256 + threadIdx.x;
  if (i < n) p[i] = 0;
}

__global__ void moe_select_kernel(const __bf16* __restrict__ h2, const __bf16* __restrict__ WselT,
                                  float* __restrict__ gates, int* __restrict__ flags) {
  int wave = threadIdx.x >> 5, lane = threadIdx.x & 31;
  int t = blockIdx.x * 8 + wave;
  int e = lane & 15, half = lane >> 4;
  const __bf16* hr = h2 + (size_t)t * DD + half * (DD / 2);
  const __bf16* ws = WselT + (size_t)e * DD + half * (DD / 2);
  float dot = 0.f;
  for (int k = 0; k < DD / 2; k += 2) {
    BF2 th, tw;
    th.u32 = *reinterpret_cast<const uint32_t*>(hr + k);
    tw.u32 = *reinterpret_cast<const uint32_t*>(ws + k);
    dot += (float)th.b[0] * (float)tw.b[0] + (float)th.b[1] * (float)tw.b[1];
  }
  dot += __shfl_xor(dot, 16, 32);
  float sel = 1.f / (1.f + __expf(-dot));
  float v1 = sel; int i1 = e;
#pragma unroll
  for (int m = 1; m < 16; m <<= 1) {
    float ov = __shfl_xor(v1, m, 32); int oi = __shfl_xor(i1, m, 32);
    if (ov > v1 || (ov == v1 && oi < i1)) { v1 = ov; i1 = oi; }
  }
  float v2 = (e == i1) ? -1.f : sel; int i2 = e;
#pragma unroll
  for (int m = 1; m < 16; m <<= 1) {
    float ov = __shfl_xor(v2, m, 32); int oi = __shfl_xor(i2, m, 32);
    if (ov > v2 || (ov == v2 && oi < i2)) { v2 = ov; i2 = oi; }
  }
  float gate = (e == i1) ? v1 : ((e == i2) ? v2 : 0.f);
  if (half == 0) {
    gates[(size_t)t * EE + e] = gate;
    if (gate > 0.f) atomicOr(&flags[(t >> 5) * EE + e], 1);
  }
}

// ---------------------------------------------------------------------------
// MoE up: up[t,e,f] = relu(h2 @ W1[e]) * gate; 32-token tile skip by flag
// grid = (T/32, E), block = 128; wave covers 64 of F=256
// ---------------------------------------------------------------------------
__global__ void moe_up_kernel(const __bf16* __restrict__ h2, const __bf16* __restrict__ W1T,
                              const float* __restrict__ gates, const int* __restrict__ flags,
                              __bf16* __restrict__ up) {
  int tt = blockIdx.x, e = blockIdx.y;
  if (!flags[tt * EE + e]) return;
  int lane = threadIdx.x & 31, wave = threadIdx.x >> 5;
  int t0 = tt * 32;
  int f0 = wave * 64;
  const __bf16* w = W1T + (size_t)e * DD * FF;  // [F][D]
  v8f acc[2][4] = {};
  gemm_tile_32x64(h2, DD, w, DD, t0, f0, 0, DD, lane, acc);
  int half = lane >> 4, ln = lane & 15;
#pragma unroll
  for (int mi = 0; mi < 2; ++mi)
#pragma unroll
    for (int ni = 0; ni < 4; ++ni) {
      int col = f0 + ni * 16 + ln;
#pragma unroll
      for (int r = 0; r < 8; ++r) {
        int t = t0 + mi * 16 + 8 * half + r;
        float g = gates[(size_t)t * EE + e];
        up[(size_t)t * (EE * FF) + e * FF + col] = (__bf16)(fmaxf(acc[mi][ni][r], 0.f) * g);
      }
    }
}

// ---------------------------------------------------------------------------
// MoE down + residual: out[t,d] = x2[t,d] + sum_e up[t,e,:] @ W2[e]
// grid = (T/32, D/256), block = 128; inactive experts skipped per tile
// ---------------------------------------------------------------------------
__global__ void moe_down_kernel(const __bf16* __restrict__ up, const __bf16* __restrict__ W2T,
                                const float* __restrict__ x2, const int* __restrict__ flags,
                                float* __restrict__ out) {
  int tt = blockIdx.x;
  int lane = threadIdx.x & 31, wave = threadIdx.x >> 5;
  int t0 = tt * 32;
  int d0 = blockIdx.y * 256 + wave * 64;
  v8f acc[2][4] = {};
  for (int e = 0; e < EE; ++e) {
    if (!flags[tt * EE + e]) continue;
    const __bf16* w = W2T + (size_t)e * DD * FF;  // [D][F]
    // A columns e*FF .. e*FF+FF of up (lda = E*F); B over K = F
    for (int k0 = 0; k0 < FF; k0 += 32) {
      v16bf a0 = load_a_frag(up, EE * FF, t0,      e * FF + k0, lane);
      v16bf a1 = load_a_frag(up, EE * FF, t0 + 16, e * FF + k0, lane);
      v16bf b0 = load_bt_frag(w, FF, d0,      k0, lane);
      v16bf b1 = load_bt_frag(w, FF, d0 + 16, k0, lane);
      v16bf b2 = load_bt_frag(w, FF, d0 + 32, k0, lane);
      v16bf b3 = load_bt_frag(w, FF, d0 + 48, k0, lane);
      acc[0][0] = wmma_bf16(a0, b0, acc[0][0]);
      acc[0][1] = wmma_bf16(a0, b1, acc[0][1]);
      acc[0][2] = wmma_bf16(a0, b2, acc[0][2]);
      acc[0][3] = wmma_bf16(a0, b3, acc[0][3]);
      acc[1][0] = wmma_bf16(a1, b0, acc[1][0]);
      acc[1][1] = wmma_bf16(a1, b1, acc[1][1]);
      acc[1][2] = wmma_bf16(a1, b2, acc[1][2]);
      acc[1][3] = wmma_bf16(a1, b3, acc[1][3]);
    }
  }
  int half = lane >> 4, ln = lane & 15;
#pragma unroll
  for (int mi = 0; mi < 2; ++mi)
#pragma unroll
    for (int ni = 0; ni < 4; ++ni) {
      int col = d0 + ni * 16 + ln;
#pragma unroll
      for (int r = 0; r < 8; ++r) {
        int t = t0 + mi * 16 + 8 * half + r;
        out[(size_t)t * DD + col] = x2[(size_t)t * DD + col] + acc[mi][ni][r];
      }
    }
}

// ---------------------------------------------------------------------------
// Host launcher
// ---------------------------------------------------------------------------
extern "C" void kernel_launch(void* const* d_in, const int* in_sizes, int n_in,
                              void* d_out, int out_size, void* d_ws, size_t ws_size,
                              hipStream_t stream) {
  (void)in_sizes; (void)n_in; (void)out_size; (void)ws_size;
  const float* x    = (const float*)d_in[0];
  const float* Wq   = (const float*)d_in[1];
  const float* Wk   = (const float*)d_in[2];
  const float* Wv   = (const float*)d_in[3];
  const float* Wo   = (const float*)d_in[4];
  const float* g1   = (const float*)d_in[5];
  const float* b1   = (const float*)d_in[6];
  const float* g2   = (const float*)d_in[7];
  const float* b2   = (const float*)d_in[8];
  const float* Wsel = (const float*)d_in[9];
  const float* W1   = (const float*)d_in[10];
  const float* W2   = (const float*)d_in[11];
  float* out = (float*)d_out;

  char* wp = (char*)d_ws;
  auto alloc = [&](size_t bytes) -> void* {
    void* r = (void*)wp;
    wp += (bytes + 255) & ~(size_t)255;
    return r;
  };
  __bf16* hbf   = (__bf16*)alloc((size_t)TT * DD * 2);
  __bf16* qb    = (__bf16*)alloc((size_t)TT * DD * 2);
  __bf16* kb    = (__bf16*)alloc((size_t)TT * DD * 2);
  __bf16* vTb   = (__bf16*)alloc((size_t)TT * DD * 2);   // [B][H][DH][S]
  __bf16* ctxb  = (__bf16*)alloc((size_t)TT * DD * 2);
  float*  x2    = (float*) alloc((size_t)TT * DD * 4);
  __bf16* h2    = (__bf16*)alloc((size_t)TT * DD * 2);
  __bf16* up    = (__bf16*)alloc((size_t)TT * EE * FF * 2);
  float*  gts   = (float*) alloc((size_t)TT * EE * 4);
  int*    flg   = (int*)   alloc((size_t)(TT / 32) * EE * 4);
  __bf16* WqT   = (__bf16*)alloc((size_t)DD * DD * 2);
  __bf16* WkT   = (__bf16*)alloc((size_t)DD * DD * 2);
  __bf16* WvT   = (__bf16*)alloc((size_t)DD * DD * 2);
  __bf16* WoT   = (__bf16*)alloc((size_t)DD * DD * 2);
  __bf16* WselT = (__bf16*)alloc((size_t)DD * EE * 2);
  __bf16* W1T   = (__bf16*)alloc((size_t)EE * DD * FF * 2);
  __bf16* W2T   = (__bf16*)alloc((size_t)EE * DD * FF * 2);

  // --- weight casts (fp32 -> bf16, transposed so K is contiguous) ---
  long long nw = (long long)DD * DD;
  tcast_kernel<<<(unsigned)((nw + 255) / 256), 256, 0, stream>>>(Wq, WqT, DD, DD, nw);
  tcast_kernel<<<(unsigned)((nw + 255) / 256), 256, 0, stream>>>(Wk, WkT, DD, DD, nw);
  tcast_kernel<<<(unsigned)((nw + 255) / 256), 256, 0, stream>>>(Wv, WvT, DD, DD, nw);
  tcast_kernel<<<(unsigned)((nw + 255) / 256), 256, 0, stream>>>(Wo, WoT, DD, DD, nw);
  long long ns = (long long)DD * EE;
  tcast_kernel<<<(unsigned)((ns + 255) / 256), 256, 0, stream>>>(Wsel, WselT, DD, EE, ns);
  long long n1 = (long long)EE * DD * FF;
  tcast_kernel<<<(unsigned)((n1 + 255) / 256), 256, 0, stream>>>(W1, W1T, DD, FF, n1);
  tcast_kernel<<<(unsigned)((n1 + 255) / 256), 256, 0, stream>>>(W2, W2T, FF, DD, n1);

  // --- attention sublayer ---
  ln_bf16_kernel<<<TT, 256, 0, stream>>>(x, g1, b1, hbf);
  dim3 gProj(TT / 32, DD / 256);
  gemm_bf16_out_kernel<<<gProj, 128, 0, stream>>>(hbf, WqT, qb, DD, DD, 0.125f); // 1/sqrt(64)
  gemm_bf16_out_kernel<<<gProj, 128, 0, stream>>>(hbf, WkT, kb, DD, DD, 1.0f);
  gemm_vT_kernel<<<gProj, 128, 0, stream>>>(hbf, WvT, vTb, DD, DD);
  dim3 gAttn(SS / 64, BB * HH);
  attn_kernel<<<gAttn, 128, 0, stream>>>(qb, kb, vTb, ctxb);
  gemm_res_f32_kernel<<<gProj, 128, 0, stream>>>(ctxb, WoT, x, x2, DD, DD);

  // --- SigmaMoE sublayer ---
  ln_bf16_kernel<<<TT, 256, 0, stream>>>(x2, g2, b2, h2);
  int nflag = (TT / 32) * EE;
  zero_i32_kernel<<<(nflag + 255) / 256, 256, 0, stream>>>(flg, nflag);
  moe_select_kernel<<<TT / 8, 256, 0, stream>>>(h2, WselT, gts, flg);
  dim3 gUp(TT / 32, EE);
  moe_up_kernel<<<gUp, 128, 0, stream>>>(h2, W1T, gts, flg, up);
  dim3 gDown(TT / 32, DD / 256);
  moe_down_kernel<<<gDown, 128, 0, stream>>>(up, W2T, x2, flg, out);
}
